// TriangleMultiplication_25924422599138
// MI455X (gfx1250) — matrix operand order repaired and checked by reading, not compiled, not run on hardware
//
#include <hip/hip_runtime.h>

// Triangle multiplicative update (outgoing), B=1, N=512, D=H=128.
// bf16 WMMA (v_wmma_f32_16x16x32_bf16) for all matmuls, fp32 layernorms.
// k2 einsum uses the CDNA5 Tensor Data Mover (tensor_load_to_lds) with
// double-buffered LDS panels and s_wait_tensorcnt pipelining.

#define NN     512
#define N2     262144      // 512*512
#define DIM    128

typedef __attribute__((ext_vector_type(16))) __bf16 v16bf;
typedef __attribute__((ext_vector_type(8)))  float  v8f;
typedef __attribute__((ext_vector_type(4)))  unsigned int v4u;
typedef __attribute__((ext_vector_type(8)))  int    v8i;
typedef __attribute__((ext_vector_type(4)))  int    v4i;

union FragBF { v16bf v; uint4 q[2]; };

__device__ __forceinline__ unsigned short f2bf(float x) {
  unsigned int u = __float_as_uint(x);
  u += 0x7FFFu + ((u >> 16) & 1u);          // round-to-nearest-even
  return (unsigned short)(u >> 16);
}
__device__ __forceinline__ float bf2f(unsigned short s) {
  return __uint_as_float(((unsigned int)s) << 16);
}
__device__ __forceinline__ float sigmoid_(float x) {
  return 1.0f / (1.0f + __expf(-x));
}

// A-matrix fragment (16x32 bf16) from row-major [m][k] panel.
// ISA layout: lane m=L%16, h=L/16; elems 0..7 <- K=h*8+0..7, elems 8..15 <- K=16+h*8+0..7
__device__ __forceinline__ v16bf frag_a(const unsigned short* base, int stride, int lane) {
  const unsigned short* p = base + (lane & 15) * stride + ((lane >> 4) << 3);
  FragBF f;
  f.q[0] = *(const uint4*)(p);
  f.q[1] = *(const uint4*)(p + 16);
  return f.v;
}
// B-matrix fragment (32x16 bf16) from row-major [n][k] panel (i.e. B^T rows).
// ISA layout: lane n=L%16, h=L/16; elems 0..7 <- K=h*8+0..7, elems 8..15 <- K=16+h*8+0..7
__device__ __forceinline__ v16bf frag_b(const unsigned short* base, int stride, int lane) {
  const unsigned short* p = base + (lane & 15) * stride + ((lane >> 4) << 3);
  FragBF f;
  f.q[0] = *(const uint4*)(p);
  f.q[1] = *(const uint4*)(p + 16);
  return f.v;
}
__device__ __forceinline__ v8f wmma_bf16(v16bf a, v16bf b, v8f c) {
  return __builtin_amdgcn_wmma_f32_16x16x32_bf16(false, a, false, b, (short)0, c, false, false);
}

#if defined(__has_builtin)
#if __has_builtin(__builtin_amdgcn_tensor_load_to_lds) && __has_builtin(__builtin_amdgcn_s_wait_tensorcnt)
#define USE_TDM 1
#endif
#endif
#ifndef USE_TDM
#define USE_TDM 0
#endif

#if USE_TDM
// Issue one TDM 2D tile load: 128 rows x 32 bf16 (64B), global row stride 512
// elems (1KB), into LDS with +16B padding after every 64B row (-> 80B rows,
// matching the [128][40] ushort panels). D# fields per CDNA5 ISA.
__device__ __forceinline__ void tdm_load_panel(unsigned lds_off, const void* gptr) {
  unsigned long long ga = (unsigned long long)gptr;
  v4u g0;
  g0[0] = 1u;                                            // count=1, user descriptor
  g0[1] = lds_off;                                       // lds_addr (bytes)
  g0[2] = (unsigned)(ga & 0xFFFFFFFFu);                  // global_addr[31:0]
  g0[3] = (unsigned)((ga >> 32) & 0x01FFFFFFu) | 0x80000000u;  // addr[56:32] | type=2
  v8i g1;
  g1[0] = (1 << 16) |                                    // data_size = 2 bytes
          (1 << 20) |                                    // pad_enable
          (3 << 22) |                                    // pad_interval: every 16 dwords (64B)
          (3 << 25);                                     // pad_amount: 4 dwords (16B)
  g1[1] = (int)((512u & 0xFFFFu) << 16);                 // tensor_dim0[15:0] = 512
  g1[2] = (int)((512u & 0xFFFFu) << 16);                 // dim0 hi=0 | tensor_dim1[15:0]=512
  g1[3] = (int)(32u << 16);                              // dim1 hi=0 | tile_dim0 = 32
  g1[4] = 128;                                           // tile_dim1 = 128 (tile_dim2=0)
  g1[5] = 512;                                           // tensor_dim0_stride lo32 = 512
  g1[6] = 0;                                             // stride hi | dim1_stride lo16
  g1[7] = 0;
  v4i z4 = {0, 0, 0, 0};
#if __clang_major__ >= 23
  v8i z8 = {0, 0, 0, 0, 0, 0, 0, 0};
  __builtin_amdgcn_tensor_load_to_lds(g0, g1, z4, z4, z8, 0);
#else
  __builtin_amdgcn_tensor_load_to_lds(g0, g1, z4, z4, 0);
#endif
}
__device__ __forceinline__ unsigned lds_off_of(const void* p) {
  return (unsigned)(unsigned long long)p;                // low 32 bits = LDS offset
}
#endif

// ---------------------------------------------------------------------------
// k0: convert weights fp32 -> bf16 into one concatenated [768][128] matrix:
// rows 0..255 = p_in_w, 256..511 = g_in_w, 512..639 = g_out_w, 640..767 = p_out_w
// ---------------------------------------------------------------------------
__global__ __launch_bounds__(256) void k0_cvt_weights(
    const float* __restrict__ p_in, const float* __restrict__ g_in,
    const float* __restrict__ g_out, const float* __restrict__ p_out,
    unsigned short* __restrict__ Wb) {
  int idx = blockIdx.x * 256 + threadIdx.x;   // 768*128 = 98304 total
  if (idx >= 98304) return;
  float v;
  if (idx < 32768)       v = p_in[idx];
  else if (idx < 65536)  v = g_in[idx - 32768];
  else if (idx < 81920)  v = g_out[idx - 65536];
  else                   v = p_out[idx - 81920];
  Wb[idx] = f2bf(v);
}

// ---------------------------------------------------------------------------
// k1: fused LayerNorm(D) + gated input projections + mask + transpose-store.
// Each block: 16 flat positions p = i*512 + k (16 consecutive k, fixed i).
//   aT[d][i][k] = sigmoid(x g_in^T)[:,d]     * (x p_in^T)[:,d]     * mask
//   bT[d][i][k] = sigmoid(x g_in^T)[:,128+d] * (x p_in^T)[:,128+d] * mask
//   gate[p][d]  = sigmoid(x g_out^T)
// ---------------------------------------------------------------------------
__global__ __launch_bounds__(256) void k1_ln_proj(
    const float* __restrict__ x_in, const float* __restrict__ mask,
    const float* __restrict__ lnw, const float* __restrict__ lnb,
    const unsigned short* __restrict__ Wb,
    unsigned short* __restrict__ aT, unsigned short* __restrict__ bT,
    unsigned short* __restrict__ gate) {
  __shared__ unsigned short xb[16][136];          // normalized rows, bf16
  __shared__ unsigned short stage[8][16][24];     // per-wave 16x16 transpose tile

  const int t   = threadIdx.x;
  const int p0  = blockIdx.x * 16;
  const int i   = p0 >> 9;
  const int k0p = p0 & 511;

  // ---- load + layernorm (fp32), 16 threads per row, 8 elems each ----
  {
    int r = t >> 4, c = t & 15;
    const float* row = x_in + (size_t)(p0 + r) * DIM + c * 8;
    float v[8];
    float s = 0.f;
#pragma unroll
    for (int e = 0; e < 8; ++e) { v[e] = row[e]; s += v[e]; }
#pragma unroll
    for (int m = 8; m >= 1; m >>= 1) s += __shfl_xor(s, m, 32);
    float mu = s * (1.f / 128.f);
    float q = 0.f;
#pragma unroll
    for (int e = 0; e < 8; ++e) { float d0 = v[e] - mu; q += d0 * d0; }
#pragma unroll
    for (int m = 8; m >= 1; m >>= 1) q += __shfl_xor(q, m, 32);
    float rstd = rsqrtf(q * (1.f / 128.f) + 1e-5f);
#pragma unroll
    for (int e = 0; e < 8; ++e) {
      int d = c * 8 + e;
      xb[r][d] = f2bf((v[e] - mu) * rstd * lnw[d] + lnb[d]);
    }
  }
  __syncthreads();

  const int w = t >> 5, lane = t & 31;
  const int h = lane >> 4, n = lane & 15;

  float m8[8];
#pragma unroll
  for (int e = 0; e < 8; ++e) m8[e] = mask[p0 + e + 8 * h];

  const v8f vzero = {0.f, 0.f, 0.f, 0.f, 0.f, 0.f, 0.f, 0.f};

  // ---- each wave: two gated p/g column tiles (a or b halves) ----
#pragma unroll
  for (int pi = 0; pi < 2; ++pi) {
    int pt = 2 * w + pi;                          // 0..15 -> a (0..7), b (8..15)
    const unsigned short* wp = Wb + (size_t)(pt * 16) * DIM;          // p_in rows
    const unsigned short* wg = Wb + (size_t)(256 + pt * 16) * DIM;    // g_in rows
    v8f accp = vzero, accg = vzero;
#pragma unroll
    for (int kk = 0; kk < DIM; kk += 32) {
      v16bf af = frag_a(&xb[0][0] + kk, 136, lane);
      accp = wmma_bf16(af, frag_b(wp + kk, DIM, lane), accp);
      accg = wmma_bf16(af, frag_b(wg + kk, DIM, lane), accg);
    }
    // gate * value * mask -> bf16, transpose via per-wave LDS tile
#pragma unroll
    for (int e = 0; e < 8; ++e) {
      int row = e + 8 * h;
      float val = sigmoid_(accg[e]) * accp[e] * m8[e];
      stage[w][n][row] = f2bf(val);               // stage[d][k]
    }
    __builtin_amdgcn_wave_barrier();              // same-wave LDS ops are in-order
    {
      int dr = lane >> 1, part = lane & 1;        // 16 d-rows x 2 x 16B
      uint4 tv = *(const uint4*)&stage[w][dr][part * 8];
      unsigned short* dst = (pt < 8) ? aT : bT;
      int d0 = (pt & 7) * 16;
      size_t ga = ((size_t)(d0 + dr) * NN + i) * NN + k0p + part * 8;
      *(uint4*)(dst + ga) = tv;
    }
    __builtin_amdgcn_wave_barrier();
  }

  // ---- each wave: one g_out tile -> gate[p][d] ----
  {
    const unsigned short* wgo = Wb + (size_t)(512 + w * 16) * DIM;
    v8f acc = vzero;
#pragma unroll
    for (int kk = 0; kk < DIM; kk += 32) {
      v16bf af = frag_a(&xb[0][0] + kk, 136, lane);
      acc = wmma_bf16(af, frag_b(wgo + kk, DIM, lane), acc);
    }
#pragma unroll
    for (int e = 0; e < 8; ++e) {
      int row = e + 8 * h;
      gate[(size_t)(p0 + row) * DIM + w * 16 + n] = f2bf(sigmoid_(acc[e]));
    }
  }
}

// ---------------------------------------------------------------------------
// k2: per-channel GEMM  out[d][i][j] = sum_k aT[d][i][k] * bT[d][j][k]
// Block = one d, 128x128 output tile. K in 32-chunks, double-buffered LDS
// panels filled by the Tensor Data Mover (wave 0 drives the DMA; compute
// waves only do ds_load + wmma). aT+bT (134 MB bf16) are L2-resident (192 MB).
// ---------------------------------------------------------------------------
__global__ __launch_bounds__(256) void k2_einsum(
    const unsigned short* __restrict__ aT, const unsigned short* __restrict__ bT,
    float* __restrict__ outmm) {
  __shared__ unsigned short As[2][128][40];       // 64B row + 16B TDM pad
  __shared__ unsigned short Bs[2][128][40];

  const int t = threadIdx.x, w = t >> 5, lane = t & 31;
  const int bj = blockIdx.x, bi = blockIdx.y, d = blockIdx.z;
  const unsigned short* gA = aT + ((size_t)d * NN + bi * 128) * NN;
  const unsigned short* gB = bT + ((size_t)d * NN + bj * 128) * NN;
  const int tiBase = (w & 3) * 2, tjBase = (w >> 2) * 4;

  const v8f vzero = {0.f, 0.f, 0.f, 0.f, 0.f, 0.f, 0.f, 0.f};
  v8f acc[2][4];
#pragma unroll
  for (int x = 0; x < 2; ++x)
#pragma unroll
    for (int y = 0; y < 4; ++y) acc[x][y] = vzero;

#if USE_TDM
  if (t < 32) {                                   // wave 0: prime buffer 0
    tdm_load_panel(lds_off_of(&As[0][0][0]), gA);
    tdm_load_panel(lds_off_of(&Bs[0][0][0]), gB);
  }
#endif

  for (int kc = 0; kc < 16; ++kc) {
    const int buf = kc & 1;
#if USE_TDM
    if (t < 32) {                                 // wave 0: DMA next, wait current
      if (kc + 1 < 16) {
        tdm_load_panel(lds_off_of(&As[buf ^ 1][0][0]), gA + (kc + 1) * 32);
        tdm_load_panel(lds_off_of(&Bs[buf ^ 1][0][0]), gB + (kc + 1) * 32);
        __builtin_amdgcn_s_wait_tensorcnt(2);     // chunk kc complete (in-order)
      } else {
        __builtin_amdgcn_s_wait_tensorcnt(0);
      }
    }
#else
    {
      int lrow = t >> 2, unit = t & 3;            // 64 rows x 4 x 16B per pass
#pragma unroll
      for (int pass = 0; pass < 2; ++pass) {
        int row = pass * 64 + lrow;
        *(uint4*)&As[buf][row][unit * 8] =
            *(const uint4*)(gA + (size_t)row * NN + kc * 32 + unit * 8);
        *(uint4*)&Bs[buf][row][unit * 8] =
            *(const uint4*)(gB + (size_t)row * NN + kc * 32 + unit * 8);
      }
    }
#endif
    __syncthreads();

    v16bf afr[2], bfr[4];
#pragma unroll
    for (int x = 0; x < 2; ++x) afr[x] = frag_a(&As[buf][(tiBase + x) * 16][0], 40, lane);
#pragma unroll
    for (int y = 0; y < 4; ++y) bfr[y] = frag_b(&Bs[buf][(tjBase + y) * 16][0], 40, lane);
#pragma unroll
    for (int x = 0; x < 2; ++x)
#pragma unroll
      for (int y = 0; y < 4; ++y)
        acc[x][y] = wmma_bf16(afr[x], bfr[y], acc[x][y]);
    __syncthreads();
  }

  const int h = lane >> 4, n = lane & 15;
  const size_t obase = (size_t)d * N2;
#pragma unroll
  for (int x = 0; x < 2; ++x)
#pragma unroll
    for (int y = 0; y < 4; ++y) {
      int gi0 = bi * 128 + (tiBase + x) * 16 + 8 * h;
      int gj  = bj * 128 + (tjBase + y) * 16 + n;
#pragma unroll
      for (int e = 0; e < 8; ++e)
        outmm[obase + (size_t)(gi0 + e) * NN + gj] = acc[x][y][e];
    }
}

// ---------------------------------------------------------------------------
// k3: LayerNorm over d + output projection (WMMA) + output gate.
// Block = 16 flat positions. outmm is [d][i][j] (d-major).
// ---------------------------------------------------------------------------
__global__ __launch_bounds__(256) void k3_ln_out(
    const float* __restrict__ outmm, const unsigned short* __restrict__ gate,
    const float* __restrict__ lnw, const float* __restrict__ lnb,
    const unsigned short* __restrict__ Wb, float* __restrict__ out) {
  __shared__ float ps[16][17];
  __shared__ float ps2[16][17];
  __shared__ float mu_s[16], rs_s[16];
  __shared__ unsigned short nb[16][136];

  const int t = threadIdx.x;
  const int p0 = blockIdx.x * 16;
  const int c = t & 15, dg = t >> 4;              // column, d-group

  float v[8];
  float s = 0.f, q = 0.f;
#pragma unroll
  for (int e = 0; e < 8; ++e) {
    int d = dg * 8 + e;
    float x = outmm[(size_t)d * N2 + p0 + c];
    v[e] = x; s += x; q += x * x;
  }
  ps[dg][c] = s;
  ps2[dg][c] = q;
  __syncthreads();
  if (t < 16) {
    float S = 0.f, Q = 0.f;
#pragma unroll
    for (int j = 0; j < 16; ++j) { S += ps[j][t]; Q += ps2[j][t]; }
    float mu = S * (1.f / 128.f);
    float var = Q * (1.f / 128.f) - mu * mu;
    mu_s[t] = mu;
    rs_s[t] = rsqrtf(var + 1e-5f);
  }
  __syncthreads();
  float mu = mu_s[c], rs = rs_s[c];
#pragma unroll
  for (int e = 0; e < 8; ++e) {
    int d = dg * 8 + e;
    nb[c][d] = f2bf((v[e] - mu) * rs * lnw[d] + lnb[d]);
  }
  __syncthreads();

  const int w = t >> 5, lane = t & 31;
  const unsigned short* wpo = Wb + (size_t)(640 + w * 16) * DIM;  // p_out rows
  const v8f vzero = {0.f, 0.f, 0.f, 0.f, 0.f, 0.f, 0.f, 0.f};
  v8f acc = vzero;
#pragma unroll
  for (int kk = 0; kk < DIM; kk += 32)
    acc = wmma_bf16(frag_a(&nb[0][0] + kk, 136, lane), frag_b(wpo + kk, DIM, lane), acc);

  const int h = lane >> 4, n = lane & 15;
#pragma unroll
  for (int e = 0; e < 8; ++e) {
    int row = e + 8 * h;
    size_t idx = (size_t)(p0 + row) * DIM + w * 16 + n;
    out[idx] = bf2f(gate[idx]) * acc[e];
  }
}

// ---------------------------------------------------------------------------
extern "C" void kernel_launch(void* const* d_in, const int* in_sizes, int n_in,
                              void* d_out, int out_size, void* d_ws, size_t ws_size,
                              hipStream_t stream) {
  const float* pair_rep   = (const float*)d_in[0];
  const float* mask       = (const float*)d_in[1];
  const float* norm_in_w  = (const float*)d_in[2];
  const float* norm_in_b  = (const float*)d_in[3];
  const float* p_in_w     = (const float*)d_in[4];
  const float* g_in_w     = (const float*)d_in[5];
  const float* norm_out_w = (const float*)d_in[6];
  const float* norm_out_b = (const float*)d_in[7];
  const float* p_out_w    = (const float*)d_in[8];
  const float* g_out_w    = (const float*)d_in[9];

  char* ws = (char*)d_ws;
  const size_t SZ_BF = (size_t)N2 * DIM * 2;      // 67,108,864 B
  unsigned short* Wb   = (unsigned short*)(ws);                        // 196,608 B
  unsigned short* aT   = (unsigned short*)(ws + 262144);               // [128][512][512] bf16
  unsigned short* bT   = (unsigned short*)(ws + 262144 + SZ_BF);
  unsigned short* gate = (unsigned short*)(ws + 262144 + 2 * SZ_BF);   // [N2][128] bf16
  float*          outmm = (float*)(ws + 262144 + 3 * SZ_BF);           // [128][512][512] f32
  float* outp = (float*)d_out;

  k0_cvt_weights<<<384, 256, 0, stream>>>(p_in_w, g_in_w, g_out_w, p_out_w, Wb);
  k1_ln_proj<<<N2 / 16, 256, 0, stream>>>(pair_rep, mask, norm_in_w, norm_in_b,
                                          Wb, aT, bT, gate);
  k2_einsum<<<dim3(4, 4, 128), 256, 0, stream>>>(aT, bT, outmm);
  k3_ln_out<<<N2 / 16, 256, 0, stream>>>(outmm, gate, norm_out_w, norm_out_b, Wb, outp);
}
